// GPT2Attention_44427141710283
// MI455X (gfx1250) — matrix operand order repaired and checked by reading, not compiled, not run on hardware
//
#include <hip/hip_runtime.h>
#include <hip/hip_bf16.h>

typedef __bf16 bf16;
typedef __bf16 v16bf __attribute__((ext_vector_type(16)));
typedef __bf16 v8bf  __attribute__((ext_vector_type(8)));
typedef float  v8f   __attribute__((ext_vector_type(8)));

#define WMMA_BF16(a, b, c) \
  __builtin_amdgcn_wmma_f32_16x16x32_bf16(false, (a), false, (b), (short)0, (c), false, false)

// ---- fragment loaders (contiguous-k order, same for A and B, wave32) -------------------------
// A (16x32 bf16): lane = M row (lane&15); 16 contiguous k at half*16
static __device__ __forceinline__ v16bf load_a_frag(const bf16* row, int kbase, int half) {
  v8bf lo = *(const v8bf*)(row + kbase + half * 16);
  v8bf hi = *(const v8bf*)(row + kbase + half * 16 + 8);
  return __builtin_shufflevector(lo, hi, 0,1,2,3,4,5,6,7,8,9,10,11,12,13,14,15);
}
// B (32x16 bf16): lane = N col (lane&15); 16 contiguous k at half*16 (col-major source)
static __device__ __forceinline__ v16bf load_b_frag(const bf16* col, int kbase, int half) {
  return *(const v16bf*)(col + kbase + half * 16);
}
static __device__ __forceinline__ float rmax16(float x) {
#pragma unroll
  for (int o = 8; o > 0; o >>= 1) x = fmaxf(x, __shfl_xor(x, o, 16));
  return x;
}
static __device__ __forceinline__ float rsum16(float x) {
#pragma unroll
  for (int o = 8; o > 0; o >>= 1) x += __shfl_xor(x, o, 16);
  return x;
}

// ---- prep kernels ----------------------------------------------------------
__global__ void f32_to_bf16_k(const float* __restrict__ in, bf16* __restrict__ out, int n) {
  int i = blockIdx.x * blockDim.x + threadIdx.x;
  if (i < n) out[i] = (bf16)in[i];
}

// in[K,N] row-major (fp32) -> out[N,K] bf16 (column-major B for WMMA B-frags)
__global__ void transpose_bf16_k(const float* __restrict__ in, bf16* __restrict__ out,
                                 int K, int N) {
  int i = blockIdx.x * blockDim.x + threadIdx.x;
  if (i >= K * N) return;
  int n = i / K, k = i - n * K;
  out[i] = (bf16)in[(size_t)k * N + n];
}

// qkv fp32 [B=2, S=2048, 3E=3072], interleaved per-head [q|k|v] slices of 192.
// Emit Qbf [BH, S, 64] (scaled 1/8), Kbf [BH, S, 64], VTbf [BH, 64, S].
__global__ void qkv_split_k(const float* __restrict__ qkv, bf16* __restrict__ Qb,
                            bf16* __restrict__ Kb, bf16* __restrict__ VTb) {
  int i = blockIdx.x * blockDim.x + threadIdx.x;
  if (i >= 2 * 2048 * 3072) return;
  int e3 = i % 3072;
  int bs = i / 3072;
  int s = bs % 2048, b = bs / 2048;
  int h = e3 / 192;
  int r = e3 - h * 192;
  int which = r >> 6, d = r & 63;
  float val = qkv[i];
  size_t bh = (size_t)b * 16 + h;
  if (which == 0)      Qb[(bh * 2048 + s) * 64 + d] = (bf16)(val * 0.125f);
  else if (which == 1) Kb[(bh * 2048 + s) * 64 + d] = (bf16)val;
  else                 VTb[(bh * 64 + d) * 2048 + s] = (bf16)val;
}

// ---- WMMA GEMM: C[M,N] = A[M,K] * Bt[N,K]^T + bias -------------------------
// Block = 4 waves; each wave owns a 32x64 tile, all share one 64-col B strip.
// B tile double-buffered in LDS via gfx1250 async-to-LDS (ASYNCcnt); A frags
// software-pipelined in registers. 8 WMMAs per staged k-step per wave.
__global__ void __launch_bounds__(128)
wmma_gemm_bias_k(const bf16* __restrict__ A, const bf16* __restrict__ Bt,
                 const float* __restrict__ bias, float* __restrict__ C,
                 int M, int N, int K) {
  const int lane = threadIdx.x & 31;
  const int wIn = threadIdx.x >> 5;
  const int nStrips = N >> 6;
  const int nStrip = blockIdx.x % nStrips;
  const int mBlock = blockIdx.x / nStrips;
  const int mTile = mBlock * 4 + wIn;  // 32-row tile
  const int half = lane >> 4, mr = lane & 15;

  // 64 rows x 32 k bf16, row padded to 40 elems (80B): 16B-aligned + conflict-free
  __shared__ bf16 Btile[2][64][40];

  // staging map: thread t copies 32B of row (t>>1), 32B-chunk (t&1)
  const int sr = threadIdx.x >> 1;
  const int sc = (threadIdx.x & 1) * 16;
  const bf16* srcRow = Bt + (size_t)(nStrip * 64 + sr) * K + sc;

#define STAGE_B(buf, k)                                                              \
  do {                                                                               \
    unsigned _dst = (unsigned)(uintptr_t)&Btile[(buf)][sr][sc];                      \
    unsigned long long _src = (unsigned long long)(uintptr_t)(srcRow + (k));         \
    asm volatile("global_load_async_to_lds_b128 %0, %1, off"                         \
                 :: "v"(_dst), "v"(_src) : "memory");                                \
    asm volatile("global_load_async_to_lds_b128 %0, %1, off offset:16"               \
                 :: "v"(_dst), "v"(_src) : "memory");                                \
  } while (0)

  const bf16* arow0 = A + (size_t)(mTile * 32 + mr) * K;
  const bf16* arow1 = arow0 + (size_t)16 * K;

  v8f acc00 = {}, acc01 = {}, acc02 = {}, acc03 = {};
  v8f acc10 = {}, acc11 = {}, acc12 = {}, acc13 = {};

  const int nk = K >> 5;
  STAGE_B(0, 0);
  v16bf a0 = load_a_frag(arow0, 0, half);
  v16bf a1 = load_a_frag(arow1, 0, half);

  for (int kk = 0; kk < nk; ++kk) {
    const int buf = kk & 1;
    asm volatile("s_wait_asynccnt 0x0" ::: "memory");  // my async writes to buf done
    __syncthreads();                                   // everyone's done; prev buf free
    if (kk + 1 < nk) STAGE_B(buf ^ 1, (kk + 1) * 32);  // prefetch next k-step
    v16bf a0n = a0, a1n = a1;
    if (kk + 1 < nk) {
      a0n = load_a_frag(arow0, (kk + 1) * 32, half);
      a1n = load_a_frag(arow1, (kk + 1) * 32, half);
    }

    v16bf vb0 = *(const v16bf*)&Btile[buf][ 0 + mr][half * 16];
    v16bf vb1 = *(const v16bf*)&Btile[buf][16 + mr][half * 16];
    v16bf vb2 = *(const v16bf*)&Btile[buf][32 + mr][half * 16];
    v16bf vb3 = *(const v16bf*)&Btile[buf][48 + mr][half * 16];
    acc00 = WMMA_BF16(a0, vb0, acc00);
    acc10 = WMMA_BF16(a1, vb0, acc10);
    acc01 = WMMA_BF16(a0, vb1, acc01);
    acc11 = WMMA_BF16(a1, vb1, acc11);
    acc02 = WMMA_BF16(a0, vb2, acc02);
    acc12 = WMMA_BF16(a1, vb2, acc12);
    acc03 = WMMA_BF16(a0, vb3, acc03);
    acc13 = WMMA_BF16(a1, vb3, acc13);
    a0 = a0n;
    a1 = a1n;
  }
#undef STAGE_B

  const int c0 = nStrip * 64 + mr;
  const float bi0 = bias ? bias[c0]      : 0.f;
  const float bi1 = bias ? bias[c0 + 16] : 0.f;
  const float bi2 = bias ? bias[c0 + 32] : 0.f;
  const float bi3 = bias ? bias[c0 + 48] : 0.f;
  const int row0 = mTile * 32 + half * 8;
#pragma unroll
  for (int v = 0; v < 8; ++v) {
    size_t r = (size_t)(row0 + v) * N;
    C[r + c0]      = acc00[v] + bi0;
    C[r + c0 + 16] = acc01[v] + bi1;
    C[r + c0 + 32] = acc02[v] + bi2;
    C[r + c0 + 48] = acc03[v] + bi3;
    size_t r2 = r + (size_t)16 * N;
    C[r2 + c0]      = acc10[v] + bi0;
    C[r2 + c0 + 16] = acc11[v] + bi1;
    C[r2 + c0 + 32] = acc12[v] + bi2;
    C[r2 + c0 + 48] = acc13[v] + bi3;
  }
}

// ---- Flash attention: wave = (b, h, 16-row q block); keys in blocks of 32 --
// K fragments pipelined across key blocks; V loads hoisted ahead of softmax.
__global__ void flash_attn_k(const bf16* __restrict__ Q, const bf16* __restrict__ Kb,
                             const bf16* __restrict__ VT, bf16* __restrict__ AO) {
  const int lane = threadIdx.x & 31;
  const int wIn = threadIdx.x >> 5;
  const int wave = blockIdx.x * 4 + wIn;
  const int qb = wave & 127;   // q block (16 rows each), S=2048
  const int bh = wave >> 7;    // 0..31 = b*16 + h
  const int b = bh >> 4, h = bh & 15;
  const int half = lane >> 4, mr = lane & 15;

  __shared__ bf16 pLds[4][16][40];  // padded stride: conflict-free + 16B aligned

  const bf16* qrow = Q + ((size_t)bh * 2048 + qb * 16 + mr) * 64;
  v16bf aq0 = load_a_frag(qrow, 0, half);
  v16bf aq1 = load_a_frag(qrow, 32, half);

  const bf16* kBase = Kb + (size_t)bh * 2048 * 64;
  const bf16* vt0 = VT + ((size_t)bh * 64 + 0  + mr) * 2048;
  const bf16* vt1 = VT + ((size_t)bh * 64 + 16 + mr) * 2048;
  const bf16* vt2 = VT + ((size_t)bh * 64 + 32 + mr) * 2048;
  const bf16* vt3 = VT + ((size_t)bh * 64 + 48 + mr) * 2048;

  v8f acc0 = {}, acc1 = {}, acc2 = {}, acc3 = {};
  float mv[8], lv[8];
#pragma unroll
  for (int v = 0; v < 8; ++v) { mv[v] = -1e30f; lv[v] = 0.f; }

  const int mrowBase = qb * 16 + half * 8;
  const int kbMax = (qb * 16 + 15) >> 5;  // inclusive; causal skip

  // prologue: K fragments for kb=0
  const bf16* kr0 = kBase + (size_t)mr * 64;
  const bf16* kr1 = kr0 + 16 * 64;
  v16bf bk00 = load_b_frag(kr0, 0, half);
  v16bf bk01 = load_b_frag(kr0, 32, half);
  v16bf bk10 = load_b_frag(kr1, 0, half);
  v16bf bk11 = load_b_frag(kr1, 32, half);

  for (int kb = 0; kb <= kbMax; ++kb) {
    const int keyBase = kb * 32;
    v8f s0 = {}, s1 = {};
    s0 = WMMA_BF16(aq0, bk00, s0);
    s0 = WMMA_BF16(aq1, bk01, s0);
    s1 = WMMA_BF16(aq0, bk10, s1);
    s1 = WMMA_BF16(aq1, bk11, s1);

    // pipeline: next key block's K fragments in flight during softmax
    v16bf nk00 = bk00, nk01 = bk01, nk10 = bk10, nk11 = bk11;
    if (kb < kbMax) {
      const bf16* nkr0 = kBase + (size_t)(keyBase + 32 + mr) * 64;
      const bf16* nkr1 = nkr0 + 16 * 64;
      nk00 = load_b_frag(nkr0, 0, half);
      nk01 = load_b_frag(nkr0, 32, half);
      nk10 = load_b_frag(nkr1, 0, half);
      nk11 = load_b_frag(nkr1, 32, half);
    }
    // V fragments for this key block, issued before the softmax ALU stretch
    v16bf bv0 = load_b_frag(vt0 + keyBase, 0, half);
    v16bf bv1 = load_b_frag(vt1 + keyBase, 0, half);
    v16bf bv2 = load_b_frag(vt2 + keyBase, 0, half);
    v16bf bv3 = load_b_frag(vt3 + keyBase, 0, half);

    const int col0 = keyBase + mr, col1 = col0 + 16;
#pragma unroll
    for (int v = 0; v < 8; ++v) {
      int mrow = mrowBase + v;
      float x0 = (col0 <= mrow) ? s0[v] : -10000.f;  // masked_bias, as in module
      float x1 = (col1 <= mrow) ? s1[v] : -10000.f;
      float rm = rmax16(fmaxf(x0, x1));
      float nm = fmaxf(mv[v], rm);
      float alpha = __expf(mv[v] - nm);
      float p0 = __expf(x0 - nm);
      float p1 = __expf(x1 - nm);
      float rs = rsum16(p0 + p1);
      lv[v] = lv[v] * alpha + rs;
      mv[v] = nm;
      acc0[v] *= alpha; acc1[v] *= alpha; acc2[v] *= alpha; acc3[v] *= alpha;
      pLds[wIn][v + 8 * half][mr]      = (bf16)p0;
      pLds[wIn][v + 8 * half][16 + mr] = (bf16)p1;
    }
    asm volatile("" ::: "memory");  // order LDS transpose (same-wave DS is in-order)
    v16bf pa = load_a_frag(&pLds[wIn][mr][0], 0, half);
    asm volatile("" ::: "memory");

    acc0 = WMMA_BF16(pa, bv0, acc0);
    acc1 = WMMA_BF16(pa, bv1, acc1);
    acc2 = WMMA_BF16(pa, bv2, acc2);
    acc3 = WMMA_BF16(pa, bv3, acc3);

    bk00 = nk00; bk01 = nk01; bk10 = nk10; bk11 = nk11;
  }

  const int rowg = qb * 16 + half * 8;
#pragma unroll
  for (int v = 0; v < 8; ++v) {
    float inv = 1.f / lv[v];
    size_t base = ((size_t)b * 2048 + rowg + v) * 1024 + h * 64 + mr;
    AO[base]      = (bf16)(acc0[v] * inv);
    AO[base + 16] = (bf16)(acc1[v] * inv);
    AO[base + 32] = (bf16)(acc2[v] * inv);
    AO[base + 48] = (bf16)(acc3[v] * inv);
  }
}

// ---- host orchestration ----------------------------------------------------
extern "C" void kernel_launch(void* const* d_in, const int* in_sizes, int n_in,
                              void* d_out, int out_size, void* d_ws, size_t ws_size,
                              hipStream_t stream) {
  (void)in_sizes; (void)n_in; (void)out_size; (void)ws_size;
  const float* hidden = (const float*)d_in[0];  // [2,2048,1024]
  const float* Wattn  = (const float*)d_in[1];  // [1024,3072]
  const float* battn  = (const float*)d_in[2];  // [3072]
  const float* Wproj  = (const float*)d_in[3];  // [1024,1024]
  const float* bproj  = (const float*)d_in[4];  // [1024]
  float* out = (float*)d_out;                   // [2,2048,1024] fp32

  char* ws = (char*)d_ws;
  bf16* Xbf    = (bf16*)ws;  ws += (size_t)4096 * 1024 * 2;
  bf16* WattnT = (bf16*)ws;  ws += (size_t)3072 * 1024 * 2;
  bf16* WprojT = (bf16*)ws;  ws += (size_t)1024 * 1024 * 2;
  float* qkv   = (float*)ws; ws += (size_t)4096 * 3072 * 4;
  bf16* Qbf    = (bf16*)ws;  ws += (size_t)32 * 2048 * 64 * 2;
  bf16* Kbf    = (bf16*)ws;  ws += (size_t)32 * 2048 * 64 * 2;
  bf16* VTbf   = (bf16*)ws;  ws += (size_t)32 * 64 * 2048 * 2;
  bf16* AObf   = (bf16*)ws;  ws += (size_t)4096 * 1024 * 2;

  // 1) convert activations + weights to bf16 (weights transposed to N-major)
  f32_to_bf16_k<<<(4096 * 1024 + 255) / 256, 256, 0, stream>>>(hidden, Xbf, 4096 * 1024);
  transpose_bf16_k<<<(1024 * 3072 + 255) / 256, 256, 0, stream>>>(Wattn, WattnT, 1024, 3072);
  transpose_bf16_k<<<(1024 * 1024 + 255) / 256, 256, 0, stream>>>(Wproj, WprojT, 1024, 1024);

  // 2) QKV projection: [4096,3072] = Xbf @ Wattn + b_attn (32-row tiles, 4 waves/block)
  wmma_gemm_bias_k<<<(4096 / 128) * (3072 / 64), 128, 0, stream>>>(
      Xbf, WattnT, battn, qkv, 4096, 3072, 1024);

  // 3) interleaved head split -> Q (pre-scaled 1/8), K, V^T in bf16
  qkv_split_k<<<(2 * 2048 * 3072 + 255) / 256, 256, 0, stream>>>(qkv, Qbf, Kbf, VTbf);

  // 4) causal flash attention -> merged-head AObf [4096,1024]
  flash_attn_k<<<(2 * 16 * 128) / 4, 128, 0, stream>>>(Qbf, Kbf, VTbf, AObf);

  // 5) output projection -> fp32 d_out
  wmma_gemm_bias_k<<<(4096 / 128) * (1024 / 64), 128, 0, stream>>>(
      AObf, WprojT, bproj, out, 4096, 1024, 1024);
}
